// GraphAttentionLayer_78142634983583
// MI455X (gfx1250) — matrix operand order repaired and checked by reading, not compiled, not run on hardware
//
#include <hip/hip_runtime.h>

#define N_NODES 50000
#define E_EDGES 800000
#define M_EDGES (E_EDGES + N_NODES)   // edges + self loops = 850000
#define F_DIM   128
#define HEADS   8
#define HEAD_DIM 16
#define G_NUM   64
#define NEG_SLOPE 0.2f
#define NORM_EPS 1e-5f

typedef __attribute__((ext_vector_type(16))) _Float16 v16h;
typedef __attribute__((ext_vector_type(8)))  _Float16 v8h;
typedef __attribute__((ext_vector_type(8)))  float    v8f;

// ---- workspace layout (float offsets) ----
#define WS_H       0                       // [N,128]  h = x@W
#define WS_ASRC    (WS_H + N_NODES*F_DIM)  // [N,8]
#define WS_ADST    (WS_ASRC + N_NODES*HEADS)
#define WS_EMAX    (WS_ADST + N_NODES*HEADS)   // unsigned (ordered-float)
#define WS_ESUM    (WS_EMAX + N_NODES*HEADS)
#define WS_GSUM    (WS_ESUM + N_NODES*HEADS)   // [64,128] -> becomes mean
#define WS_GSQ     (WS_GSUM + G_NUM*F_DIM)     // [64,128] -> becomes invstd
#define WS_CNT     (WS_GSQ + G_NUM*F_DIM)      // [64]
#define WS_WT      (WS_CNT + G_NUM)            // [128,128] f16 (8192 floats)
#define WS_ZERO_BEG WS_EMAX
#define WS_ZERO_CNT (WS_CNT + G_NUM - WS_ZERO_BEG)

// monotone float <-> ordered-uint mapping (for atomicMax on floats)
__device__ __forceinline__ unsigned f2ord(float f) {
  unsigned u = __float_as_uint(f);
  return (u & 0x80000000u) ? ~u : (u | 0x80000000u);
}
__device__ __forceinline__ float ord2f(unsigned u) {
  return (u & 0x80000000u) ? __uint_as_float(u & 0x7fffffffu) : __uint_as_float(~u);
}

// ---------------- 0: zero d_out (agg) + stats region of ws ----------------
__global__ void k_init(float* out, float* ws_zero) {
  const int total = N_NODES * F_DIM + WS_ZERO_CNT;
  for (int i = blockIdx.x * blockDim.x + threadIdx.x; i < total;
       i += gridDim.x * blockDim.x) {
    if (i < N_NODES * F_DIM) out[i] = 0.0f;
    else                     ws_zero[i - N_NODES * F_DIM] = 0.0f;
  }
}

// ------- 0b: Wt[n][k] = (f16) W[k][n]  (transpose + convert, 64KB->32KB) ---
__global__ void k_cvtW(const float* __restrict__ W, _Float16* __restrict__ Wt) {
  int t = blockIdx.x * blockDim.x + threadIdx.x;     // t = k*128 + n (coalesced read)
  if (t >= F_DIM * F_DIM) return;
  int k = t >> 7, n = t & 127;
  Wt[n * F_DIM + k] = (_Float16)W[t];
}

// ---------------- 1: h = x @ W via v_wmma_f32_16x16x32_f16 ----------------
// One wave computes rows [row0, row0+16) x all 128 cols.
// A (16x32 f16): lane m = lane&15, K = (lane>>4)*8 + {0..7, 16..23}
// B (32x16 f16): lane n = lane&15, K = (lane>>4)*8 + {0..7, 16..23} -> two runs in Wt
// C (16x16 f32): lane n = lane&15, vgpr j -> m = j + (lane>>4)*8
__global__ void k_gemm(const float* __restrict__ x, const _Float16* __restrict__ Wt,
                       float* __restrict__ h) {
  const int lane = threadIdx.x & 31;
  const int wave = blockIdx.x * (blockDim.x >> 5) + (threadIdx.x >> 5);
  const int row0 = wave * 16;
  if (row0 >= N_NODES) return;          // wave-uniform: EXEC stays all-1s
  const int lr  = lane & 15;
  const int lhi = lane >> 4;

  v8f acc[8];
#pragma unroll
  for (int t = 0; t < 8; ++t)
#pragma unroll
    for (int j = 0; j < 8; ++j) acc[t][j] = 0.0f;

  const float* xrow = x + (size_t)(row0 + lr) * F_DIM;

  for (int kb = 0; kb < F_DIM; kb += 32) {
    // ---- A fragment: two runs of 8 consecutive floats -> 4x b128 loads ----
    const int kbase = kb + (lhi << 3);
    float4 p0 = *(const float4*)(xrow + kbase + 0);
    float4 p1 = *(const float4*)(xrow + kbase + 4);
    float4 p2 = *(const float4*)(xrow + kbase + 16);
    float4 p3 = *(const float4*)(xrow + kbase + 20);
    v16h a;
    a[0]  = (_Float16)p0.x; a[1]  = (_Float16)p0.y;
    a[2]  = (_Float16)p0.z; a[3]  = (_Float16)p0.w;
    a[4]  = (_Float16)p1.x; a[5]  = (_Float16)p1.y;
    a[6]  = (_Float16)p1.z; a[7]  = (_Float16)p1.w;
    a[8]  = (_Float16)p2.x; a[9]  = (_Float16)p2.y;
    a[10] = (_Float16)p2.z; a[11] = (_Float16)p2.w;
    a[12] = (_Float16)p3.x; a[13] = (_Float16)p3.y;
    a[14] = (_Float16)p3.z; a[15] = (_Float16)p3.w;

#pragma unroll
    for (int t = 0; t < 8; ++t) {
      const int c0 = t * 16;
      // ---- B fragment: two runs of 8 f16 in Wt (K = lhi*8 + {0..7,16..23}) -> 2x b128 loads ----
      const _Float16* wp = Wt + (size_t)(c0 + lr) * F_DIM + kb + (lhi << 3);
      v8h b0 = *(const v8h*)(wp);
      v8h b1 = *(const v8h*)(wp + 16);
      v16h b;
#pragma unroll
      for (int i = 0; i < 8; ++i) { b[i] = b0[i]; b[8 + i] = b1[i]; }
      acc[t] = __builtin_amdgcn_wmma_f32_16x16x32_f16(
          false, a, false, b, (short)0, acc[t], false, false);
    }
  }
#pragma unroll
  for (int t = 0; t < 8; ++t) {
    const int c0 = t * 16;
#pragma unroll
    for (int j = 0; j < 8; ++j) {
      int m = j + (lhi << 3);
      h[(size_t)(row0 + m) * F_DIM + c0 + lr] = acc[t][j];
    }
  }
}

// ---------------- 2: per-node attention scores a_src/a_dst [N,8] ----------
__global__ void k_att(const float* __restrict__ h, const float* __restrict__ att_src,
                      const float* __restrict__ att_dst, float* __restrict__ a_src,
                      float* __restrict__ a_dst) {
  int t = blockIdx.x * blockDim.x + threadIdx.x;
  if (t >= N_NODES * HEADS) return;
  int n = t >> 3, hd = t & 7;
  const float* hp = h + (size_t)n * F_DIM + hd * HEAD_DIM;
  const float* as = att_src + hd * HEAD_DIM;
  const float* ad = att_dst + hd * HEAD_DIM;
  float s = 0.0f, d = 0.0f;
#pragma unroll
  for (int i = 0; i < HEAD_DIM; ++i) {
    float v = hp[i];
    s += v * as[i];
    d += v * ad[i];
  }
  a_src[t] = s;
  a_dst[t] = d;
}

// ---------------- 3: segment max over incoming edges -----------------------
__global__ void k_edge_max(const int* __restrict__ ei, const float* __restrict__ a_src,
                           const float* __restrict__ a_dst, unsigned* __restrict__ e_max) {
  int m = blockIdx.x * blockDim.x + threadIdx.x;
  if (m >= M_EDGES) return;
  int s = (m < E_EDGES) ? ei[m] : (m - E_EDGES);
  int d = (m < E_EDGES) ? ei[E_EDGES + m] : (m - E_EDGES);
#pragma unroll
  for (int hd = 0; hd < HEADS; ++hd) {
    float e = a_src[s * HEADS + hd] + a_dst[d * HEADS + hd];
    e = e > 0.0f ? e : NEG_SLOPE * e;
    atomicMax(&e_max[d * HEADS + hd], f2ord(e));
  }
}

// ---------------- 4: segment sum of exp(e - max) ---------------------------
__global__ void k_edge_sum(const int* __restrict__ ei, const float* __restrict__ a_src,
                           const float* __restrict__ a_dst,
                           const unsigned* __restrict__ e_max, float* __restrict__ e_sum) {
  int m = blockIdx.x * blockDim.x + threadIdx.x;
  if (m >= M_EDGES) return;
  int s = (m < E_EDGES) ? ei[m] : (m - E_EDGES);
  int d = (m < E_EDGES) ? ei[E_EDGES + m] : (m - E_EDGES);
#pragma unroll
  for (int hd = 0; hd < HEADS; ++hd) {
    float e = a_src[s * HEADS + hd] + a_dst[d * HEADS + hd];
    e = e > 0.0f ? e : NEG_SLOPE * e;
    float ex = __expf(e - ord2f(e_max[d * HEADS + hd]));
    atomicAdd(&e_sum[d * HEADS + hd], ex);
  }
}

// ---------------- 5: weighted aggregation (one wave32 per edge) ------------
__global__ void k_edge_agg(const int* __restrict__ ei, const float* __restrict__ h,
                           const float* __restrict__ a_src, const float* __restrict__ a_dst,
                           const unsigned* __restrict__ e_max, const float* __restrict__ e_sum,
                           float* __restrict__ agg) {
  const int lane = threadIdx.x & 31;
  const int m = blockIdx.x * (blockDim.x >> 5) + (threadIdx.x >> 5);
  if (m >= M_EDGES) return;                  // wave-uniform
  int s = (m < E_EDGES) ? ei[m] : (m - E_EDGES);
  int d = (m < E_EDGES) ? ei[E_EDGES + m] : (m - E_EDGES);
  // every lane computes alpha for head = lane&7 (4x redundant, wave32-friendly)
  int hd = lane & 7;
  float e = a_src[s * HEADS + hd] + a_dst[d * HEADS + hd];
  e = e > 0.0f ? e : NEG_SLOPE * e;
  float alpha = __expf(e - ord2f(e_max[d * HEADS + hd])) /
                (e_sum[d * HEADS + hd] + 1e-16f);
  const float* hs = h + (size_t)s * F_DIM;
  float* ap = agg + (size_t)d * F_DIM;
#pragma unroll
  for (int j = 0; j < 4; ++j) {
    int f = lane + 32 * j;                   // feature index 0..127
    float a = __shfl(alpha, f >> 4, 32);     // head = f/16 lives in lane (f>>4)
    atomicAdd(&ap[f], hs[f] * a);
  }
}

// ---------------- 6: bias + residual, per-graph sums -----------------------
__global__ void k_post1(float* __restrict__ out, const float* __restrict__ x,
                        const float* __restrict__ bias, const int* __restrict__ batch,
                        float* __restrict__ g_sum, float* __restrict__ counts) {
  int t = blockIdx.x * blockDim.x + threadIdx.x;
  if (t >= N_NODES * F_DIM) return;
  int n = t >> 7, f = t & 127;
  float v = out[t] + bias[f] + x[t];
  out[t] = v;
  int g = batch[n];
  atomicAdd(&g_sum[g * F_DIM + f], v);
  if (f == 0) atomicAdd(&counts[g], 1.0f);
}

// ---------------- 7: mean = sum / max(count,1) -----------------------------
__global__ void k_mean(float* __restrict__ g_sum, const float* __restrict__ counts) {
  int i = blockIdx.x * blockDim.x + threadIdx.x;
  if (i >= G_NUM * F_DIM) return;
  float c = counts[i >> 7];
  c = c < 1.0f ? 1.0f : c;
  g_sum[i] /= c;
}

// ---------------- 8: sub = out - mean_scale*mean; accumulate sub^2 ---------
__global__ void k_post2(float* __restrict__ out, const float* __restrict__ mean,
                        const float* __restrict__ mean_scale, const int* __restrict__ batch,
                        float* __restrict__ g_sq) {
  int t = blockIdx.x * blockDim.x + threadIdx.x;
  if (t >= N_NODES * F_DIM) return;
  int n = t >> 7, f = t & 127;
  int g = batch[n];
  float sub = out[t] - mean_scale[f] * mean[g * F_DIM + f];
  out[t] = sub;
  atomicAdd(&g_sq[g * F_DIM + f], sub * sub);
}

// ---------------- 9: invstd = rsqrt(var + eps) -----------------------------
__global__ void k_invstd(float* __restrict__ g_sq, const float* __restrict__ counts) {
  int i = blockIdx.x * blockDim.x + threadIdx.x;
  if (i >= G_NUM * F_DIM) return;
  float c = counts[i >> 7];
  c = c < 1.0f ? 1.0f : c;
  g_sq[i] = rsqrtf(g_sq[i] / c + NORM_EPS);
}

// ---------------- 10: final affine -----------------------------------------
__global__ void k_post3(float* __restrict__ out, const float* __restrict__ invstd,
                        const float* __restrict__ nw, const float* __restrict__ nb,
                        const int* __restrict__ batch) {
  int t = blockIdx.x * blockDim.x + threadIdx.x;
  if (t >= N_NODES * F_DIM) return;
  int n = t >> 7, f = t & 127;
  int g = batch[n];
  out[t] = nw[f] * out[t] * invstd[g * F_DIM + f] + nb[f];
}

extern "C" void kernel_launch(void* const* d_in, const int* in_sizes, int n_in,
                              void* d_out, int out_size, void* d_ws, size_t ws_size,
                              hipStream_t stream) {
  const float* x          = (const float*)d_in[0];
  const int*   edge_index = (const int*)d_in[1];
  const int*   batch      = (const int*)d_in[2];
  const float* W_att      = (const float*)d_in[3];
  const float* att_src    = (const float*)d_in[4];
  const float* att_dst    = (const float*)d_in[5];
  const float* bias_att   = (const float*)d_in[6];
  const float* norm_w     = (const float*)d_in[7];
  const float* norm_b     = (const float*)d_in[8];
  const float* mean_scale = (const float*)d_in[9];
  float* out = (float*)d_out;

  float*     ws     = (float*)d_ws;
  float*     h      = ws + WS_H;
  float*     a_src  = ws + WS_ASRC;
  float*     a_dst  = ws + WS_ADST;
  unsigned*  e_max  = (unsigned*)(ws + WS_EMAX);
  float*     e_sum  = ws + WS_ESUM;
  float*     g_sum  = ws + WS_GSUM;
  float*     g_sq   = ws + WS_GSQ;
  float*     counts = ws + WS_CNT;
  _Float16*  Wt     = (_Float16*)(ws + WS_WT);

  // 0: zero agg (d_out) + stats; convert/transpose W to f16
  k_init<<<2048, 256, 0, stream>>>(out, ws + WS_ZERO_BEG);
  k_cvtW<<<(F_DIM * F_DIM + 255) / 256, 256, 0, stream>>>(W_att, Wt);

  // 1: h = x @ W  (WMMA f16 -> f32). 3125 waves of 16 rows, 4 waves/block.
  {
    int waves = (N_NODES + 15) / 16;
    int blocks = (waves + 3) / 4;
    k_gemm<<<blocks, 128, 0, stream>>>(x, Wt, h);
  }
  // 2: attention scores
  k_att<<<(N_NODES * HEADS + 255) / 256, 256, 0, stream>>>(h, att_src, att_dst, a_src, a_dst);
  // 3: segment max
  k_edge_max<<<(M_EDGES + 255) / 256, 256, 0, stream>>>(edge_index, a_src, a_dst, e_max);
  // 4: segment sum of exp
  k_edge_sum<<<(M_EDGES + 255) / 256, 256, 0, stream>>>(edge_index, a_src, a_dst, e_max, e_sum);
  // 5: aggregation, one wave per edge (8 waves / 256-thread block)
  k_edge_agg<<<(M_EDGES + 7) / 8, 256, 0, stream>>>(edge_index, h, a_src, a_dst, e_max, e_sum, out);
  // 6-10: bias/residual + GraphNorm
  int nf_blocks = (N_NODES * F_DIM + 255) / 256;
  k_post1<<<nf_blocks, 256, 0, stream>>>(out, x, bias_att, batch, g_sum, counts);
  k_mean<<<(G_NUM * F_DIM + 63) / 64, 64, 0, stream>>>(g_sum, counts);
  k_post2<<<nf_blocks, 256, 0, stream>>>(out, g_sum, mean_scale, batch, g_sq);
  k_invstd<<<(G_NUM * F_DIM + 63) / 64, 64, 0, stream>>>(g_sq, counts);
  k_post3<<<nf_blocks, 256, 0, stream>>>(out, g_sq, norm_w, norm_b, batch);
}
